// EncoderLSTM_4320737099990
// MI455X (gfx1250) — matrix operand order repaired and checked by reading, not compiled, not run on hardware
//
#include <hip/hip_runtime.h>
#include <hip/hip_bf16.h>
#include <stdint.h>

#define B_  128
#define T_  512
#define F_  1024
#define H_  1024
#define N4_ 4096

typedef __attribute__((ext_vector_type(16))) __bf16 v16bf;
typedef __attribute__((ext_vector_type(8)))  float  v8f;

union FragU {
  uint4 q[2];
  v16bf v;
};

// ---------------------------------------------------------- async LDS staging
// CDNA5 GLOBAL_LOAD_ASYNC_TO_LDS_B128 (ASYNCcnt) when available, else a
// synchronous global_load_b128 + ds_store_b128 copy.
#define HAS_ASYNC_LDS __has_builtin(__builtin_amdgcn_global_load_async_to_lds_b128)

#if HAS_ASYNC_LDS
typedef int async_i32x4 __attribute__((vector_size(4 * sizeof(int))));
typedef __attribute__((address_space(1))) async_i32x4* async_g128p;
typedef __attribute__((address_space(3))) async_i32x4* async_l128p;
#endif

__device__ __forceinline__ void stage16(const void* g, void* l) {
#if HAS_ASYNC_LDS
  __builtin_amdgcn_global_load_async_to_lds_b128(
      (async_g128p)(uintptr_t)g,
      // generic LDS address carries the DS byte offset in its low 32 bits
      (async_l128p)(unsigned)(uintptr_t)l,
      0, 0);
#else
  *(uint4*)l = *(const uint4*)g;
#endif
}

__device__ __forceinline__ void stage_wait() {
#if HAS_ASYNC_LDS
#if __has_builtin(__builtin_amdgcn_s_wait_asynccnt)
  __builtin_amdgcn_s_wait_asynccnt(0);
#else
  asm volatile("s_wait_asynccnt 0x0" ::: "memory");
#endif
#endif
}

// ------------------------------------------------------------- WMMA fragments
// A 16x32 bf16 fragment from LDS tile AT[row][k], row stride 40 elements.
// lane<16: M=lane, K=0..7 (q0), 16..23 (q1); lane>=16: M=lane-16, K=8..15, 24..31
__device__ __forceinline__ v16bf load_frag_a_lds(const unsigned short* AT, int lane) {
  const int row  = lane & 15;
  const int koff = (lane & 16) ? 8 : 0;
  const unsigned short* p = AT + row * 40 + koff;
  FragU f;
  f.q[0] = *(const uint4*)(p);
  f.q[1] = *(const uint4*)(p + 16);
  return f.v;
}

// B 32x16 bf16 fragment from LDS tile BT[col][k], row stride 40 elements.
// lane<16: N=lane, K=0..7, 16..23 ; lane>=16: N=lane-16, K=8..15, 24..31
__device__ __forceinline__ v16bf load_frag_b_lds(const unsigned short* BT, int lane) {
  const int col  = lane & 15;
  const int koff = (lane & 16) ? 8 : 0;
  const unsigned short* p = BT + col * 40 + koff;
  FragU f;
  f.q[0] = *(const uint4*)(p);
  f.q[1] = *(const uint4*)(p + 16);
  return f.v;
}

__device__ __forceinline__ v8f wmma_bf16(v16bf a, v16bf b, v8f c) {
  return __builtin_amdgcn_wmma_f32_16x16x32_bf16(false, a, false, b, (short)0, c,
                                                 false, false);
}

__device__ __forceinline__ unsigned short f32_to_bf16_rne(float x) {
  unsigned u = __float_as_uint(x);
  return (unsigned short)((u + 0x7FFFu + ((u >> 16) & 1u)) >> 16);
}

// ---------------------------------------------------------------- prep kernels

__global__ void f32_to_bf16_kernel(const float* __restrict__ in,
                                   unsigned short* __restrict__ out, int n) {
  int i = blockIdx.x * blockDim.x + threadIdx.x;
  for (; i < n; i += gridDim.x * blockDim.x) out[i] = f32_to_bf16_rne(in[i]);
}

// W (K x N fp32, row-major) -> WT (N x K bf16, row-major). 32x32 LDS tiles.
__global__ __launch_bounds__(256) void transpose_to_bf16_kernel(
    const float* __restrict__ W, unsigned short* __restrict__ WT, int K, int N) {
  __shared__ unsigned short tile[32][33];
  const int n0 = blockIdx.x * 32;
  const int k0 = blockIdx.y * 32;
  const int c  = threadIdx.x & 31;
  const int r4 = threadIdx.x >> 5;
#pragma unroll
  for (int i = 0; i < 4; ++i) {
    int r = r4 + i * 8;
    tile[r][c] = f32_to_bf16_rne(W[(size_t)(k0 + r) * N + n0 + c]);
  }
  __syncthreads();
  const int kx = threadIdx.x & 31;
  const int nr = threadIdx.x >> 5;
#pragma unroll
  for (int i = 0; i < 4; ++i) {
    int n = nr + i * 8;
    WT[(size_t)(n0 + n) * K + k0 + kx] = tile[kx][n];
  }
}

__global__ void zero_u32_kernel(unsigned int* __restrict__ p, int n) {
  int i = blockIdx.x * blockDim.x + threadIdx.x;
  for (; i < n; i += gridDim.x * blockDim.x) p[i] = 0u;
}

// Exclusive prefix-OR of (x[b,t,0] != 0) over t, per batch row.
__global__ void eos_prefix_kernel(const float* __restrict__ x,
                                  unsigned char* __restrict__ mask) {
  int b = threadIdx.x;
  if (b >= B_) return;
  unsigned char flag = 0;
  const float* xb = x + (size_t)b * T_ * F_;
  for (int t = 0; t < T_; ++t) {
    mask[b * T_ + t] = flag;
    if (xb[(size_t)t * F_] != 0.0f) flag = 1;
  }
}

// ------------------------------------------------- big GEMM: Zx = Xbf @ Wi_bf
// A: M x K bf16 row-major; WT: N x K bf16 row-major (pre-transposed weights).
// Block tile 64(M) x 128(N), 8 waves (2x4), wave = 32x32 (2x2 WMMA tiles).
// Double-buffered LDS staging via async copy.
__global__ __launch_bounds__(256) void gemm_xwi_kernel(
    const unsigned short* __restrict__ A,
    const unsigned short* __restrict__ WT,
    float* __restrict__ D, int M, int K, int N) {
  __shared__ unsigned short Asm[2][64 * 40];
  __shared__ unsigned short Bsm[2][128 * 40];
  const int tid = threadIdx.x;
  const int wave = tid >> 5, lane = tid & 31;
  const int m0 = blockIdx.y * 64;
  const int n0 = blockIdx.x * 128;
  const int wm = (wave >> 2) * 32;
  const int wn = (wave & 3) * 32;

  // copy-role indices: A tile 64 rows x 32 k, B tile 128 cols x 32 k
  const int arow = tid >> 2, ach = tid & 3;   // 8-elem (16 B) chunk of a row
  const int bcol = tid >> 1, bch = tid & 1;   // 16-elem (32 B) chunk of a col
  const unsigned short* ga0 = A  + (size_t)(m0 + arow) * K + ach * 8;
  const unsigned short* gb0 = WT + (size_t)(n0 + bcol) * K + bch * 16;

  v8f acc00 = {}, acc01 = {}, acc10 = {}, acc11 = {};

  auto stage_tile = [&](int buf, int kk) {
    stage16(ga0 + kk, &Asm[buf][arow * 40 + ach * 8]);
    stage16(gb0 + kk, &Bsm[buf][bcol * 40 + bch * 16]);
    stage16(gb0 + kk + 8, &Bsm[buf][bcol * 40 + bch * 16 + 8]);
  };

  stage_tile(0, 0);
  int buf = 0;
  for (int kk = 0; kk < K; kk += 32) {
    stage_wait();
    __syncthreads();
    if (kk + 32 < K) stage_tile(buf ^ 1, kk + 32);
    v16bf a0 = load_frag_a_lds(&Asm[buf][wm * 40], lane);
    v16bf a1 = load_frag_a_lds(&Asm[buf][(wm + 16) * 40], lane);
    v16bf b0 = load_frag_b_lds(&Bsm[buf][wn * 40], lane);
    v16bf b1 = load_frag_b_lds(&Bsm[buf][(wn + 16) * 40], lane);
    acc00 = wmma_bf16(a0, b0, acc00);
    acc01 = wmma_bf16(a0, b1, acc01);
    acc10 = wmma_bf16(a1, b0, acc10);
    acc11 = wmma_bf16(a1, b1, acc11);
    buf ^= 1;
  }
  const int mloc = (lane >> 4) << 3;
  const int nloc = lane & 15;
#pragma unroll
  for (int r = 0; r < 8; ++r) {
    int m = mloc + r;
    D[(size_t)(m0 + wm + m) * N + n0 + wn + nloc]           = acc00[r];
    D[(size_t)(m0 + wm + m) * N + n0 + wn + 16 + nloc]      = acc01[r];
    D[(size_t)(m0 + wm + 16 + m) * N + n0 + wn + nloc]      = acc10[r];
    D[(size_t)(m0 + wm + 16 + m) * N + n0 + wn + 16 + nloc] = acc11[r];
  }
}

// --------------------------------------- fused recurrent step: z=h@Wh, gates
// grid: 4 batch-tiles x 32 j-tiles = 128 blocks, 256 threads (8 waves).
// wave w: gate = w/2, 16-column half = w&1; wave computes 32(M) x 16(N).
__global__ __launch_bounds__(256) void lstm_step_kernel(
    const unsigned short* __restrict__ h_in,   // B x H bf16 (carry in)
    unsigned short* __restrict__ h_out,        // B x H bf16 (carry out)
    const unsigned short* __restrict__ WhT,    // 4H x H bf16 (pre-transposed)
    const float* __restrict__ Zx,              // (B*T) x 4H fp32 preactivations
    const float* __restrict__ bias,            // 4H fp32
    float* __restrict__ c_io,                  // B x H fp32 carry
    const unsigned char* __restrict__ eos,     // B x T exclusive prefix mask
    float* __restrict__ y,                     // B x T x H fp32 output
    int t) {
  __shared__ unsigned short Asm[2][32 * 40];
  __shared__ unsigned short Bsm[2][128 * 40];
  __shared__ float zs[32 * 132];
  const int tid = threadIdx.x;
  const int wave = tid >> 5, lane = tid & 31;
  const int bt = blockIdx.x >> 5;   // batch tile (32 rows)
  const int jt = blockIdx.x & 31;   // hidden tile (32 cols)
  const int gate = wave >> 1;
  const int ncl = (gate << 5) | ((wave & 1) << 4);  // local z column 0..112

  // copy roles: A tile 32 rows x 32 k (threads 0..127), B tile 128 cols x 32 k
  const int arow = (tid >> 2) & 31, ach = tid & 3;
  const int bcol = tid >> 1, bch = tid & 1;
  const int bn = (bcol >> 5) * 1024 + jt * 32 + (bcol & 31);  // WT row (gate strip)
  const unsigned short* ga0 = h_in + (size_t)(bt * 32 + arow) * H_ + ach * 8;
  const unsigned short* gb0 = WhT + (size_t)bn * H_ + bch * 16;

  v8f acc0 = {}, acc1 = {};

  auto stage_tile = [&](int buf, int kk) {
    if (tid < 128) stage16(ga0 + kk, &Asm[buf][arow * 40 + ach * 8]);
    stage16(gb0 + kk, &Bsm[buf][bcol * 40 + bch * 16]);
    stage16(gb0 + kk + 8, &Bsm[buf][bcol * 40 + bch * 16 + 8]);
  };

  stage_tile(0, 0);
  int buf = 0;
  for (int kk = 0; kk < H_; kk += 32) {
    stage_wait();
    __syncthreads();
    if (kk + 32 < H_) stage_tile(buf ^ 1, kk + 32);
    v16bf a0 = load_frag_a_lds(&Asm[buf][0], lane);
    v16bf a1 = load_frag_a_lds(&Asm[buf][16 * 40], lane);
    v16bf bf = load_frag_b_lds(&Bsm[buf][ncl * 40], lane);
    acc0 = wmma_bf16(a0, bf, acc0);
    acc1 = wmma_bf16(a1, bf, acc1);
    buf ^= 1;
  }
  const int mloc = (lane >> 4) << 3;
  const int nloc = lane & 15;
#pragma unroll
  for (int r = 0; r < 8; ++r) {
    zs[(mloc + r) * 132 + ncl + nloc]      = acc0[r];
    zs[(mloc + r + 16) * 132 + ncl + nloc] = acc1[r];
  }
  __syncthreads();

  for (int e = tid; e < 1024; e += 256) {
    int m = e >> 5, j = e & 31;
    int bg = bt * 32 + m;
    int jg = jt * 32 + j;
    size_t zxo = ((size_t)bg * T_ + t) * N4_;
    float zi = zs[m * 132 + j]      + Zx[zxo + jg]        + bias[jg];
    float zf = zs[m * 132 + 32 + j] + Zx[zxo + 1024 + jg] + bias[1024 + jg];
    float zg = zs[m * 132 + 64 + j] + Zx[zxo + 2048 + jg] + bias[2048 + jg];
    float zo = zs[m * 132 + 96 + j] + Zx[zxo + 3072 + jg] + bias[3072 + jg];
    float ig = 1.0f / (1.0f + __expf(-zi));
    float fg = 1.0f / (1.0f + __expf(-zf));
    float gg = tanhf(zg);
    float og = 1.0f / (1.0f + __expf(-zo));
    size_t ci = (size_t)bg * H_ + jg;
    float cp = c_io[ci];
    float cn = fg * cp + ig * gg;
    float hn = og * tanhf(cn);
    // scan output is the UNMASKED new_h; only the carry freezes after EOS
    y[((size_t)bg * T_ + t) * H_ + jg] = hn;
    if (eos[(size_t)bg * T_ + t]) {
      h_out[ci] = h_in[ci];          // frozen carry (ping-pong copy)
    } else {
      c_io[ci]  = cn;
      h_out[ci] = f32_to_bf16_rne(hn);
    }
  }
}

// ----------------------------------------------------------------- host side

extern "C" void kernel_launch(void* const* d_in, const int* in_sizes, int n_in,
                              void* d_out, int out_size, void* d_ws, size_t ws_size,
                              hipStream_t stream) {
  const float* x    = (const float*)d_in[0];
  const float* Wi   = (const float*)d_in[1];
  const float* Wh   = (const float*)d_in[2];
  const float* bias = (const float*)d_in[3];
  float* y = (float*)d_out;
  (void)in_sizes; (void)n_in; (void)out_size; (void)ws_size;

  char* ws = (char*)d_ws;
  size_t off = 0;
  auto alloc = [&](size_t bytes) -> void* {
    void* p = (void*)(ws + off);
    off += (bytes + 255) & ~(size_t)255;
    return p;
  };
  float*          Zx   = (float*)alloc((size_t)B_ * T_ * N4_ * 4);  // 1 GiB
  unsigned short* xbf  = (unsigned short*)alloc((size_t)B_ * T_ * F_ * 2);
  unsigned short* WiT  = (unsigned short*)alloc((size_t)F_ * N4_ * 2);
  unsigned short* WhT  = (unsigned short*)alloc((size_t)H_ * N4_ * 2);
  unsigned short* h0   = (unsigned short*)alloc((size_t)B_ * H_ * 2);
  unsigned short* h1   = (unsigned short*)alloc((size_t)B_ * H_ * 2);
  float*          cbuf = (float*)alloc((size_t)B_ * H_ * 4);
  unsigned char*  eos  = (unsigned char*)alloc((size_t)B_ * T_);

  // precision cast, weight pre-transposition (once), state init
  f32_to_bf16_kernel<<<8192, 256, 0, stream>>>(x, xbf, B_ * T_ * F_);
  transpose_to_bf16_kernel<<<dim3(N4_ / 32, F_ / 32), 256, 0, stream>>>(Wi, WiT, F_, N4_);
  transpose_to_bf16_kernel<<<dim3(N4_ / 32, H_ / 32), 256, 0, stream>>>(Wh, WhT, H_, N4_);
  zero_u32_kernel<<<256, 256, 0, stream>>>((unsigned int*)h0, B_ * H_ / 2);
  zero_u32_kernel<<<256, 256, 0, stream>>>((unsigned int*)h1, B_ * H_ / 2);
  zero_u32_kernel<<<256, 256, 0, stream>>>((unsigned int*)cbuf, B_ * H_);
  eos_prefix_kernel<<<1, 128, 0, stream>>>(x, eos);

  // hoisted input projection: Zx = X @ Wi for all timesteps (parallel WMMA)
  gemm_xwi_kernel<<<dim3(N4_ / 128, (B_ * T_) / 64), 256, 0, stream>>>(
      xbf, WiT, Zx, B_ * T_, F_, N4_);

  // sequential recurrence, one fused kernel per step, h carry ping-pong
  for (int t = 0; t < T_; ++t) {
    const unsigned short* hin = (t & 1) ? h1 : h0;
    unsigned short*      hout = (t & 1) ? h0 : h1;
    lstm_step_kernel<<<128, 256, 0, stream>>>(hin, hout, WhT, Zx, bias, cbuf,
                                              eos, y, t);
  }
}
